// PointNetFeaturePropagation_23149873725518
// MI455X (gfx1250) — matrix operand order repaired and checked by reading, not compiled, not run on hardware
//
#include <hip/hip_runtime.h>
#include <stdint.h>

// ---------------- problem constants (fixed by the reference) ----------------
static constexpr int Bn   = 4;
static constexpr int Np   = 16384;   // N points (xyz1)
static constexpr int NPp  = 4096;    // npoint (xyz2)
static constexpr int C1   = 128;
static constexpr int C2   = 256;
static constexpr int CIN  = C1 + C2; // 384
static constexpr int H1c  = 256;
static constexpr int H2c  = 256;
static constexpr int COUT = 128;
static constexpr int BN   = Bn * Np; // 65536 rows through the MLP

// ---------------- types for WMMA ----------------
typedef __attribute__((ext_vector_type(16))) __bf16 v16bf;
typedef __attribute__((ext_vector_type(8)))  float  v8f;
typedef int v4i __attribute__((vector_size(16)));

__device__ __forceinline__ unsigned short f2bf(float f) {
  unsigned u = __float_as_uint(f);
  u += 0x7FFFu + ((u >> 16) & 1u);   // round-to-nearest-even
  return (unsigned short)(u >> 16);
}

// ---------------- CDNA5 async global->LDS copy (ASYNCcnt path) --------------
#if __has_builtin(__builtin_amdgcn_global_load_async_to_lds_b128)
#define ASYNC_VIA_BUILTIN 1
#else
#define ASYNC_VIA_BUILTIN 0
#endif

__device__ __forceinline__ void async_copy16(const void* g, void* l) {
#if ASYNC_VIA_BUILTIN
  __builtin_amdgcn_global_load_async_to_lds_b128(
      (__attribute__((address_space(1))) v4i*)(g),
      (__attribute__((address_space(3))) v4i*)(l), 0, 0);
#else
  // GV-mode: vdst = LDS byte address, vaddr = 64-bit global address
  unsigned lds_off =
      (unsigned)(unsigned long long)((__attribute__((address_space(3))) char*)l);
  asm volatile("global_load_async_to_lds_b128 %0, %1, off"
               :
               : "v"(lds_off), "v"(g)
               : "memory");
#endif
}

__device__ __forceinline__ void async_wait0() {
#if __has_builtin(__builtin_amdgcn_s_wait_asynccnt)
  __builtin_amdgcn_s_wait_asynccnt(0);
#else
  asm volatile("s_wait_asynccnt 0x0" ::: "memory");
#endif
}

// ---------------------------------------------------------------------------
// Kernel 1: weight convert fp32->bf16 with transpose: Wt[n*K + k] = W[k*N + n]
// ---------------------------------------------------------------------------
__global__ __launch_bounds__(256) void transpose_bf16_kernel(
    const float* __restrict__ W, unsigned short* __restrict__ Wt, int K, int Nc) {
  int i = blockIdx.x * 256 + threadIdx.x;
  if (i >= K * Nc) return;
  int k = i / Nc, n = i - k * Nc;
  Wt[(size_t)n * K + k] = f2bf(W[i]);
}

// ---------------------------------------------------------------------------
// Kernel 2: three_nn. xyz2 for one batch (48KB) staged to LDS via async copies.
// grid = (Np/256, B), block = 256.
// ---------------------------------------------------------------------------
__global__ __launch_bounds__(256) void three_nn_kernel(
    const float* __restrict__ xyz1, const float* __restrict__ xyz2,
    int* __restrict__ idx_out, float* __restrict__ w_out) {
  __shared__ __align__(16) float s2[NPp * 3];   // 48 KB
  const int b = blockIdx.y;
  const float* src = xyz2 + (size_t)b * NPp * 3;
  for (int i = threadIdx.x; i < (NPp * 3) / 4; i += 256)
    async_copy16(src + i * 4, s2 + i * 4);
  async_wait0();
  __syncthreads();

  const int n = blockIdx.x * 256 + threadIdx.x;
  const float* p = xyz1 + ((size_t)b * Np + n) * 3;
  const float px = p[0], py = p[1], pz = p[2];

  float d0 = 3e38f, d1 = 3e38f, d2 = 3e38f;
  int   i0 = 0, i1 = 0, i2 = 0;
  for (int j = 0; j < NPp; ++j) {            // uniform LDS address -> broadcast
    const float dx = px - s2[j * 3 + 0];
    const float dy = py - s2[j * 3 + 1];
    const float dz = pz - s2[j * 3 + 2];
    const float d = dx * dx + dy * dy + dz * dz;
    if (d < d2) {
      if (d < d0)      { d2 = d1; i2 = i1; d1 = d0; i1 = i0; d0 = d; i0 = j; }
      else if (d < d1) { d2 = d1; i2 = i1; d1 = d;  i1 = j; }
      else             { d2 = d;  i2 = j; }
    }
  }
  d0 = fmaxf(d0, 1e-10f); d1 = fmaxf(d1, 1e-10f); d2 = fmaxf(d2, 1e-10f);
  const float r0 = 1.0f / d0, r1 = 1.0f / d1, r2 = 1.0f / d2;
  const float rs = 1.0f / (r0 + r1 + r2);
  const size_t o = ((size_t)b * Np + n) * 3;
  idx_out[o] = i0; idx_out[o + 1] = i1; idx_out[o + 2] = i2;
  w_out[o] = r0 * rs; w_out[o + 1] = r1 * rs; w_out[o + 2] = r2 * rs;
}

// ---------------------------------------------------------------------------
// Kernel 3: three_interpolate + concat -> X0 bf16 [BN][384].
// One wave per point: idx/weights are wave-uniform, channel reads coalesced.
// ---------------------------------------------------------------------------
__global__ __launch_bounds__(256) void interp_concat_kernel(
    const float* __restrict__ f1, const float* __restrict__ f2,
    const int* __restrict__ idx, const float* __restrict__ w,
    unsigned short* __restrict__ X0) {
  const int wave = threadIdx.x >> 5, lane = threadIdx.x & 31;
  const int p = blockIdx.x * 8 + wave;       // 0..BN-1
  const int b = p >> 14;                     // Np = 16384
  const size_t o = (size_t)p * 3;
  const int i0 = idx[o], i1 = idx[o + 1], i2 = idx[o + 2];
  const float w0 = w[o], w1 = w[o + 1], w2 = w[o + 2];
  const float* r0 = f2 + ((size_t)b * NPp + i0) * C2;
  const float* r1 = f2 + ((size_t)b * NPp + i1) * C2;
  const float* r2 = f2 + ((size_t)b * NPp + i2) * C2;
  const float* fin = f1 + (size_t)p * C1;
  unsigned short* out = X0 + (size_t)p * CIN;
  for (int c = lane; c < C1; c += 32) out[c] = f2bf(fin[c]);
  for (int c = lane; c < C2; c += 32)
    out[C1 + c] = f2bf(w0 * r0[c] + w1 * r1[c] + w2 * r2[c]);
}

// ---------------------------------------------------------------------------
// Kernels 4/5: bf16 WMMA GEMM: Out[M][N] = act(A[M][K] @ W[K][N] + bias)
// A row-major bf16, Wt stored transposed [N][K] bf16.
// Block = 256 thr (8 waves): block tile 128(M) x 128(N); wave tile 32x64 ->
// 8 WMMA accumulators per wave. K stepped by 32 through double-buffered LDS,
// filled with GLOBAL_LOAD_ASYNC_TO_LDS_B128; one s_wait_asynccnt+barrier/step.
// ---------------------------------------------------------------------------
template <bool RELU, bool OUTBF>
__global__ __launch_bounds__(256) void gemm_wmma_kernel(
    const unsigned short* __restrict__ A, const unsigned short* __restrict__ Wt,
    const float* __restrict__ bias, void* __restrict__ Out,
    int M, int Nc, int K) {
  __shared__ __align__(16) unsigned short sA[2][128 * 32]; // 2 x 8 KB
  __shared__ __align__(16) unsigned short sB[2][128 * 32]; // 2 x 8 KB

  const int tid  = threadIdx.x;
  const int wave = tid >> 5, lane = tid & 31;
  const int wm = wave & 3, wn = wave >> 2;   // 4(M) x 2(N) wave grid
  const int l16 = lane & 15, hi = lane >> 4;
  const int bm = blockIdx.x * 128, bn = blockIdx.y * 128;

  v8f acc[2][4] = {};

  // per-thread staging coordinates: 512 chunks of 8 halves per tile, 2/thread
  const int ar0 = tid >> 2,         ac0 = (tid & 3) * 8;
  const int ar1 = (tid + 256) >> 2, ac1 = ((tid + 256) & 3) * 8;

  auto stage = [&](int buf, int k0) {
    async_copy16(&A [(size_t)(bm + ar0) * K + k0 + ac0], &sA[buf][ar0 * 32 + ac0]);
    async_copy16(&A [(size_t)(bm + ar1) * K + k0 + ac1], &sA[buf][ar1 * 32 + ac1]);
    async_copy16(&Wt[(size_t)(bn + ar0) * K + k0 + ac0], &sB[buf][ar0 * 32 + ac0]);
    async_copy16(&Wt[(size_t)(bn + ar1) * K + k0 + ac1], &sB[buf][ar1 * 32 + ac1]);
  };

  const int nk = K >> 5;
  stage(0, 0);
  async_wait0();
  __syncthreads();

  union Frag { v16bf v; uint4 q[2]; };

  for (int t = 0; t < nk; ++t) {
    const int cur = t & 1;
    if (t + 1 < nk) stage(cur ^ 1, (t + 1) << 5);   // prefetch next tile (async)

    Frag a[2], b[4];
#pragma unroll
    for (int mi = 0; mi < 2; ++mi) {
      const unsigned short* pa = &sA[cur][(wm * 32 + mi * 16 + l16) * 32];
      // A layout (ISA 7.12.2): elems 0..7 -> k=8*hi.., elems 8..15 -> k=16+8*hi..
      a[mi].q[0] = *(const uint4*)(pa + 8 * hi);
      a[mi].q[1] = *(const uint4*)(pa + 16 + 8 * hi);
    }
#pragma unroll
    for (int ni = 0; ni < 4; ++ni) {
      const unsigned short* pb = &sB[cur][(wn * 64 + ni * 16 + l16) * 32];
      // B layout: elems 0..7 -> k=8*hi.., elems 8..15 -> k=16+8*hi.. (N-major)
      b[ni].q[0] = *(const uint4*)(pb + 8 * hi);
      b[ni].q[1] = *(const uint4*)(pb + 16 + 8 * hi);
    }
#pragma unroll
    for (int mi = 0; mi < 2; ++mi)
#pragma unroll
      for (int ni = 0; ni < 4; ++ni)
        acc[mi][ni] = __builtin_amdgcn_wmma_f32_16x16x32_bf16(
            false, a[mi].v, false, b[ni].v, (short)0, acc[mi][ni], false, false);

    async_wait0();      // next tile fully landed in LDS
    __syncthreads();    // all waves done reading cur / writing next
  }

  // ---- epilogue: C/D layout: lane -> n = l16, VGPR r -> m = r + 8*hi ----
#pragma unroll
  for (int mi = 0; mi < 2; ++mi)
#pragma unroll
    for (int ni = 0; ni < 4; ++ni) {
      const int gn = bn + wn * 64 + ni * 16 + l16;
      const float bv = bias[gn];
#pragma unroll
      for (int r = 0; r < 8; ++r) {
        const int gm = bm + wm * 32 + mi * 16 + 8 * hi + r;
        float v = acc[mi][ni][r] + bv;
        if (RELU) v = fmaxf(v, 0.0f);
        if (OUTBF)
          ((unsigned short*)Out)[(size_t)gm * Nc + gn] = f2bf(v);
        else
          ((float*)Out)[(size_t)gm * Nc + gn] = v;
      }
    }
}

// ---------------------------------------------------------------------------
extern "C" void kernel_launch(void* const* d_in, const int* in_sizes, int n_in,
                              void* d_out, int out_size, void* d_ws, size_t ws_size,
                              hipStream_t stream) {
  (void)in_sizes; (void)n_in; (void)out_size; (void)ws_size;
  const float* xyz1 = (const float*)d_in[0];
  const float* xyz2 = (const float*)d_in[1];
  const float* f1   = (const float*)d_in[2];
  const float* f2   = (const float*)d_in[3];
  const float* W1   = (const float*)d_in[4];
  const float* b1   = (const float*)d_in[5];
  const float* W2   = (const float*)d_in[6];
  const float* b2   = (const float*)d_in[7];
  const float* W3   = (const float*)d_in[8];
  const float* b3   = (const float*)d_in[9];

  // workspace layout (256B aligned)
  char* ws = (char*)d_ws;
  size_t off = 0;
  auto take = [&](size_t bytes) {
    void* p = ws + off;
    off += (bytes + 255) & ~(size_t)255;
    return p;
  };
  int*            idxb = (int*)           take((size_t)BN * 3 * 4);
  float*          wb   = (float*)         take((size_t)BN * 3 * 4);
  unsigned short* X0   = (unsigned short*)take((size_t)BN * CIN * 2); // 50.3 MB
  unsigned short* Hb1  = (unsigned short*)take((size_t)BN * H1c * 2); // 33.6 MB
  unsigned short* Wt1  = (unsigned short*)take((size_t)CIN * H1c * 2);
  unsigned short* Wt2  = (unsigned short*)take((size_t)H1c * H2c * 2);
  unsigned short* Wt3  = (unsigned short*)take((size_t)H2c * COUT * 2);
  unsigned short* Hb2  = X0;  // reuse: X0 is dead after GEMM1 reads it

  // 1) weights -> bf16, transposed to [N][K]
  transpose_bf16_kernel<<<(CIN * H1c + 255) / 256, 256, 0, stream>>>(W1, Wt1, CIN, H1c);
  transpose_bf16_kernel<<<(H1c * H2c + 255) / 256, 256, 0, stream>>>(W2, Wt2, H1c, H2c);
  transpose_bf16_kernel<<<(H2c * COUT + 255) / 256, 256, 0, stream>>>(W3, Wt3, H2c, COUT);

  // 2) three_nn (xyz2 batch resident in LDS)
  three_nn_kernel<<<dim3(Np / 256, Bn), 256, 0, stream>>>(xyz1, xyz2, idxb, wb);

  // 3) interpolate + concat -> X0 (bf16)
  interp_concat_kernel<<<BN / 8, 256, 0, stream>>>(f1, f2, idxb, wb, X0);

  // 4) MLP via WMMA GEMMs (block tile 128x128)
  gemm_wmma_kernel<true, true><<<dim3(BN / 128, H1c / 128), 256, 0, stream>>>(
      X0, Wt1, b1, Hb1, BN, H1c, CIN);
  gemm_wmma_kernel<true, true><<<dim3(BN / 128, H2c / 128), 256, 0, stream>>>(
      Hb1, Wt2, b2, Hb2, BN, H2c, H1c);
  gemm_wmma_kernel<false, false><<<dim3(BN / 128, COUT / 128), 256, 0, stream>>>(
      Hb2, Wt3, b3, d_out, BN, COUT, H2c);
}
